// SeqRegressor_69879117906672
// MI455X (gfx1250) — matrix operand order repaired and checked by reading, not compiled, not run on hardware
//
#include <hip/hip_runtime.h>
#include <hip/hip_bf16.h>
#include <math.h>

#define NTOKEN 32000
#define NEMB   512
#define NHID   1024
#define NATT   1024
#define NOUT   512
#define TGTLEN 64
#define BS     32
#define SRCLEN 128

typedef __attribute__((ext_vector_type(16))) __bf16        v16bf;
typedef __attribute__((ext_vector_type(8)))  float         v8f;
typedef __attribute__((ext_vector_type(4)))  unsigned int  v4u;

union FragBF { v16bf v; v4u q[2]; };

__device__ __forceinline__ float bf2f(unsigned short s) {
  return __uint_as_float(((unsigned int)s) << 16);
}
__device__ __forceinline__ unsigned short f2bf(float f) {
  unsigned int u = __float_as_uint(f);
  u += 0x7FFFu + ((u >> 16) & 1u);          // round-to-nearest-even
  return (unsigned short)(u >> 16);
}
__device__ __forceinline__ float sigmoidf_(float x) {
  return 1.0f / (1.0f + __expf(-x));
}

// Accumulate one 16x16 output tile over klen (multiple of 32) K-elements.
// Arow: &W[mrow*ldA + col0]  (weights, row-major [M][K], lane-specific row)
// Brow: &B[nrow*ldB + col0]  (activations, row-major [N][K], lane-specific row)
// Fragment layout per CDNA5 ISA 7.12.2 (16-bit A 16x32; B mirrored over N).
__device__ __forceinline__ v8f wmma_mac(v8f acc,
    const unsigned short* __restrict__ Arow,
    const unsigned short* __restrict__ Brow,
    int klen, bool hi) {
  const int aoff = hi ? 8 : 0;    // A: lanes 16-31 hold K+8..15 / K+24..31
  const int boff = hi ? 8 : 0;    // B: lanes 16-31 hold K+8..15 / K+24..31
#pragma unroll 4
  for (int k = 0; k < klen; k += 32) {
    FragBF fa, fb;
    fa.q[0] = *(const v4u*)(Arow + k + aoff);
    fa.q[1] = *(const v4u*)(Arow + k + aoff + 16);
    fb.q[0] = *(const v4u*)(Brow + k + boff);
    fb.q[1] = *(const v4u*)(Brow + k + boff + 16);
    acc = __builtin_amdgcn_wmma_f32_16x16x32_bf16(
        false, fa.v, false, fb.v, (short)0, acc, false, false);
  }
  return acc;
}

// ---------------- setup kernels ----------------
__global__ void cvt_bf16_kernel(const float* __restrict__ src,
                                unsigned short* __restrict__ dst, int n) {
  int i = blockIdx.x * blockDim.x + threadIdx.x;
  int stride = gridDim.x * blockDim.x;
  for (; i < n; i += stride) dst[i] = f2bf(src[i]);
}

__global__ void gather_emb_kernel(const int* __restrict__ tgt,
                                  const float* __restrict__ emb,
                                  unsigned short* __restrict__ xemb) {
  int i = blockIdx.x * blockDim.x + threadIdx.x;   // TGT*BS*NEMB
  if (i >= TGTLEN * BS * NEMB) return;
  int k  = i & (NEMB - 1);
  int tb = i >> 9;                                  // t*BS + b
  int row = tgt[tb];
  xemb[i] = f2bf(emb[(size_t)row * NEMB + k]);
}

// internal state layouts: hInt/cInt are [layer][j][b] (f32), hbf is [layer][b][j] (bf16)
__global__ void init_state_kernel(const float* __restrict__ h0,
                                  const float* __restrict__ c0,
                                  float* __restrict__ hInt, float* __restrict__ cInt,
                                  unsigned short* __restrict__ hbf) {
  int i = blockIdx.x * blockDim.x + threadIdx.x;   // 2*BS*NHID
  if (i >= 2 * BS * NHID) return;
  int j = i & (NHID - 1);
  int b = (i >> 10) & (BS - 1);
  int l = i >> 15;
  float hv = h0[i], cv = c0[i];
  hInt[l * (NHID * BS) + j * BS + b] = hv;
  cInt[l * (NHID * BS) + j * BS + b] = cv;
  hbf[i] = f2bf(hv);
}

__global__ void zero_bf_kernel(unsigned short* __restrict__ p, int n) {
  int i = blockIdx.x * blockDim.x + threadIdx.x;
  if (i < n) p[i] = 0;
}

// ---------------- LSTM gate GEMM ----------------
// gates[m][n] = bih[m]+bhh[m] + sum_k Wih[m][k]*inp[n][k] + sum_k Whh[m][k]*h[n][k]
// inp is split into two K-segments (B1 cols [0,K1), B2 cols [K1,K1+K2) of Wih).
// 256 M-tiles x 2 N-tiles = 512 waves -> grid 128 x 128 threads.
__global__ void gemm_gates_kernel(
    const unsigned short* __restrict__ Wih, int ldWih,
    const unsigned short* __restrict__ Whh,            // ld = NHID
    const unsigned short* __restrict__ B1, int ldB1, int K1,
    const unsigned short* __restrict__ B2, int ldB2, int K2,
    const unsigned short* __restrict__ Hprev,          // [BS][NHID]
    const float* __restrict__ bih, const float* __restrict__ bhh,
    float* __restrict__ gates)                         // [4*NHID][BS]
{
  int tid  = threadIdx.x;
  int lane = tid & 31;
  int wave = blockIdx.x * (blockDim.x >> 5) + (tid >> 5);
  int tn = wave & 1, tm = wave >> 1;
  int lr = lane & 15;
  bool hi = lane >= 16;
  int mrow = tm * 16 + lr;
  int nrow = tn * 16 + lr;

  v8f acc = {0.f, 0.f, 0.f, 0.f, 0.f, 0.f, 0.f, 0.f};
  acc = wmma_mac(acc, Wih + (size_t)mrow * ldWih,
                      B1  + (size_t)nrow * ldB1, K1, hi);
  if (K2)
    acc = wmma_mac(acc, Wih + (size_t)mrow * ldWih + K1,
                        B2  + (size_t)nrow * ldB2, K2, hi);
  acc = wmma_mac(acc, Whh + (size_t)mrow * NHID,
                      Hprev + (size_t)nrow * NHID, NHID, hi);

  int n = tn * 16 + lr;
#pragma unroll
  for (int j = 0; j < 8; ++j) {
    int m = tm * 16 + j + (hi ? 8 : 0);
    gates[m * BS + n] = acc[j] + bih[m] + bhh[m];
  }
}

// ---------------- LSTM cell (pointwise) ----------------
__global__ void lstm_cell_kernel(
    const float* __restrict__ gates,          // [4*NHID][BS]
    float* __restrict__ hF, float* __restrict__ cF,  // [NHID][BS]
    unsigned short* __restrict__ hB,          // bf16 [BS][NHID]
    unsigned short* __restrict__ hB2, int ld2)// optional second bf16 copy
{
  int i = blockIdx.x * blockDim.x + threadIdx.x;   // BS*NHID
  if (i >= BS * NHID) return;
  int n = i & (BS - 1);
  int j = i >> 5;
  float ig = sigmoidf_(gates[j * BS + n]);
  float fg = sigmoidf_(gates[(NHID + j) * BS + n]);
  float gg = tanhf    (gates[(2 * NHID + j) * BS + n]);
  float og = sigmoidf_(gates[(3 * NHID + j) * BS + n]);
  float c = fg * cF[j * BS + n] + ig * gg;
  float h = og * tanhf(c);
  cF[j * BS + n] = c;
  hF[j * BS + n] = h;
  unsigned short hb = f2bf(h);
  hB[n * NHID + j] = hb;
  if (hB2) hB2[n * ld2 + j] = hb;
}

// ---------------- attention: scores -> softmax -> context ----------------
// one block per batch column n; 128 threads (one per source position)
__global__ void attention_kernel(
    const unsigned short* __restrict__ ctx,   // bf16 [SRC][BS][NHID]
    const float* __restrict__ h1,             // [NHID][BS] internal layout
    const unsigned char* __restrict__ mask,   // [SRC][BS]
    unsigned short* __restrict__ catAnt)      // bf16, write catAnt[n*2048 + j]
{
  __shared__ float hsh[NHID];
  __shared__ float sc[SRCLEN];
  __shared__ float red[SRCLEN];
  int n = blockIdx.x;
  int tid = threadIdx.x;

  for (int j = tid; j < NHID; j += SRCLEN) hsh[j] = h1[j * BS + n];
  __syncthreads();

  const unsigned short* crow = ctx + ((size_t)tid * BS + n) * NHID;
  float dot = 0.f;
  for (int j = 0; j < NHID; ++j) dot += bf2f(crow[j]) * hsh[j];
  if (mask[(size_t)tid * BS + n]) dot = -INFINITY;
  sc[tid] = dot;
  red[tid] = dot;
  __syncthreads();
  for (int off = SRCLEN / 2; off > 0; off >>= 1) {
    if (tid < off) red[tid] = fmaxf(red[tid], red[tid + off]);
    __syncthreads();
  }
  float mx = red[0];
  __syncthreads();
  float e = __expf(sc[tid] - mx);
  red[tid] = e;
  __syncthreads();
  for (int off = SRCLEN / 2; off > 0; off >>= 1) {
    if (tid < off) red[tid] += red[tid + off];
    __syncthreads();
  }
  float inv = 1.0f / red[0];
  __syncthreads();
  sc[tid] = e * inv;                           // prob
  __syncthreads();

  for (int j = tid; j < NHID; j += SRCLEN) {
    float a = 0.f;
    for (int s = 0; s < SRCLEN; ++s)
      a += sc[s] * bf2f(ctx[((size_t)s * BS + n) * NHID + j]);
    catAnt[n * (2 * NHID) + j] = f2bf(a);
  }
}

// ---------------- attention projection: att = tanh(cat @ Wp^T) ----------------
// 64 M-tiles x 2 N-tiles = 128 waves -> grid 32 x 128 threads.
__global__ void gemm_tanh_kernel(
    const unsigned short* __restrict__ Wp,    // [NATT][2*NHID]
    const unsigned short* __restrict__ cat,   // [BS][2*NHID]
    unsigned short* __restrict__ attOut)      // bf16 [BS][NATT]
{
  int tid  = threadIdx.x;
  int lane = tid & 31;
  int wave = blockIdx.x * (blockDim.x >> 5) + (tid >> 5);
  int tn = wave & 1, tm = wave >> 1;
  int lr = lane & 15;
  bool hi = lane >= 16;
  int mrow = tm * 16 + lr;
  int nrow = tn * 16 + lr;

  v8f acc = {0.f, 0.f, 0.f, 0.f, 0.f, 0.f, 0.f, 0.f};
  acc = wmma_mac(acc, Wp + (size_t)mrow * (2 * NHID),
                      cat + (size_t)nrow * (2 * NHID), 2 * NHID, hi);
  int n = tn * 16 + lr;
#pragma unroll
  for (int j = 0; j < 8; ++j) {
    int m = tm * 16 + j + (hi ? 8 : 0);
    attOut[n * NATT + m] = f2bf(tanhf(acc[j]));
  }
}

// ---------------- final projection: dec = att_outs @ Wr^T + br ----------------
// 32 M-tiles x 128 N-tiles = 4096 waves -> grid 1024 x 128 threads.
__global__ void gemm_out_kernel(
    const unsigned short* __restrict__ Wr,    // [NOUT][NATT]
    const unsigned short* __restrict__ Bm,    // [TGT*BS][NATT]
    const float* __restrict__ br,
    float* __restrict__ out)                  // [TGT*BS][NOUT]
{
  int tid  = threadIdx.x;
  int lane = tid & 31;
  int wave = blockIdx.x * (blockDim.x >> 5) + (tid >> 5);
  int tn = wave & 127, tm = wave >> 7;
  int lr = lane & 15;
  bool hi = lane >= 16;
  int mrow = tm * 16 + lr;
  int nrow = tn * 16 + lr;

  v8f acc = {0.f, 0.f, 0.f, 0.f, 0.f, 0.f, 0.f, 0.f};
  acc = wmma_mac(acc, Wr + (size_t)mrow * NATT,
                      Bm + (size_t)nrow * NATT, NATT, hi);
  int n = tn * 16 + lr;
#pragma unroll
  for (int j = 0; j < 8; ++j) {
    int m = tm * 16 + j + (hi ? 8 : 0);
    out[(size_t)n * NOUT + m] = acc[j] + br[m];
  }
}

// ---------------- final state transpose ----------------
__global__ void finalize_state_kernel(const float* __restrict__ hInt,
                                      const float* __restrict__ cInt,
                                      float* __restrict__ outH,
                                      float* __restrict__ outC) {
  int i = blockIdx.x * blockDim.x + threadIdx.x;   // 2*BS*NHID
  if (i >= 2 * BS * NHID) return;
  int j = i & (NHID - 1);
  int b = (i >> 10) & (BS - 1);
  int l = i >> 15;
  outH[i] = hInt[l * (NHID * BS) + j * BS + b];
  outC[i] = cInt[l * (NHID * BS) + j * BS + b];
}

// ---------------- host ----------------
extern "C" void kernel_launch(void* const* d_in, const int* in_sizes, int n_in,
                              void* d_out, int out_size, void* d_ws, size_t ws_size,
                              hipStream_t stream) {
  (void)in_sizes; (void)n_in; (void)out_size; (void)ws_size;
  const int*   tgt  = (const int*)d_in[0];
  const float* h0   = (const float*)d_in[1];
  const float* c0   = (const float*)d_in[2];
  const float* ctx  = (const float*)d_in[3];
  const unsigned char* mask = (const unsigned char*)d_in[4];
  const float* emb  = (const float*)d_in[5];
  const float* Wih0 = (const float*)d_in[6];
  const float* Whh0 = (const float*)d_in[7];
  const float* bih0 = (const float*)d_in[8];
  const float* bhh0 = (const float*)d_in[9];
  const float* Wih1 = (const float*)d_in[10];
  const float* Whh1 = (const float*)d_in[11];
  const float* bih1 = (const float*)d_in[12];
  const float* bhh1 = (const float*)d_in[13];
  const float* Wp   = (const float*)d_in[14];
  const float* Wr   = (const float*)d_in[15];
  const float* br   = (const float*)d_in[16];

  char* ws = (char*)d_ws;
  size_t off = 0;
  auto alloc = [&](size_t bytes) -> char* {
    char* p = ws + off;
    off = (off + bytes + 255) & ~(size_t)255;
    return p;
  };
  unsigned short* wih0b = (unsigned short*)alloc((size_t)4*NHID*1536*2);
  unsigned short* whh0b = (unsigned short*)alloc((size_t)4*NHID*NHID*2);
  unsigned short* wih1b = (unsigned short*)alloc((size_t)4*NHID*NHID*2);
  unsigned short* whh1b = (unsigned short*)alloc((size_t)4*NHID*NHID*2);
  unsigned short* wpb   = (unsigned short*)alloc((size_t)NATT*2*NHID*2);
  unsigned short* wrb   = (unsigned short*)alloc((size_t)NOUT*NATT*2);
  unsigned short* ctxb  = (unsigned short*)alloc((size_t)SRCLEN*BS*NHID*2);
  unsigned short* xembb = (unsigned short*)alloc((size_t)TGTLEN*BS*NEMB*2);
  unsigned short* attprev = (unsigned short*)alloc((size_t)(TGTLEN+1)*BS*NATT*2);
  float* hInt  = (float*)alloc((size_t)2*NHID*BS*4);
  float* cInt  = (float*)alloc((size_t)2*NHID*BS*4);
  unsigned short* hbf = (unsigned short*)alloc((size_t)2*BS*NHID*2);
  unsigned short* catbuf = (unsigned short*)alloc((size_t)BS*2*NHID*2);
  float* gates = (float*)alloc((size_t)4*NHID*BS*4);

  // ---- one-time conversions / init ----
  auto cvt = [&](const float* s, unsigned short* d, int n) {
    int grid = (n + 255) / 256; if (grid > 4096) grid = 4096;
    cvt_bf16_kernel<<<grid, 256, 0, stream>>>(s, d, n);
  };
  cvt(Wih0, wih0b, 4*NHID*1536);
  cvt(Whh0, whh0b, 4*NHID*NHID);
  cvt(Wih1, wih1b, 4*NHID*NHID);
  cvt(Whh1, whh1b, 4*NHID*NHID);
  cvt(Wp,   wpb,   NATT*2*NHID);
  cvt(Wr,   wrb,   NOUT*NATT);
  cvt(ctx,  ctxb,  SRCLEN*BS*NHID);
  gather_emb_kernel<<<(TGTLEN*BS*NEMB + 255)/256, 256, 0, stream>>>(tgt, emb, xembb);
  init_state_kernel<<<(2*BS*NHID + 255)/256, 256, 0, stream>>>(h0, c0, hInt, cInt, hbf);
  zero_bf_kernel<<<(BS*NATT + 255)/256, 256, 0, stream>>>(attprev, BS*NATT);

  // ---- recurrent loop ----
  for (int t = 0; t < TGTLEN; ++t) {
    const unsigned short* xemb_t = xembb + (size_t)t * BS * NEMB;
    const unsigned short* attp_t = attprev + (size_t)t * BS * NATT;
    unsigned short*       attp_n = attprev + (size_t)(t + 1) * BS * NATT;

    // layer 0: inp = [x_t | att_prev], h = hbf[0]
    gemm_gates_kernel<<<128, 128, 0, stream>>>(
        wih0b, 1536, whh0b,
        xemb_t, NEMB, NEMB,
        attp_t, NATT, NATT,
        hbf, bih0, bhh0, gates);
    lstm_cell_kernel<<<(BS*NHID + 255)/256, 256, 0, stream>>>(
        gates, hInt, cInt, hbf, (unsigned short*)nullptr, 0);

    // layer 1: inp = h0_new (hbf[0]), h = hbf[1]; h1 also goes to catbuf[:, :1024]
    gemm_gates_kernel<<<128, 128, 0, stream>>>(
        wih1b, NHID, whh1b,
        hbf, NHID, NHID,
        (const unsigned short*)nullptr, 0, 0,
        hbf + (size_t)BS * NHID, bih1, bhh1, gates);
    lstm_cell_kernel<<<(BS*NHID + 255)/256, 256, 0, stream>>>(
        gates, hInt + (size_t)NHID * BS, cInt + (size_t)NHID * BS,
        hbf + (size_t)BS * NHID, catbuf, 2 * NHID);

    // attention context -> catbuf[:, 1024:]
    attention_kernel<<<BS, SRCLEN, 0, stream>>>(
        ctxb, hInt + (size_t)NHID * BS, mask, catbuf + NHID);

    // att = tanh(cat @ Wp^T) -> attprev slot t+1
    gemm_tanh_kernel<<<32, 128, 0, stream>>>(wpb, catbuf, attp_n);
  }

  // ---- output projection + final states ----
  float* out = (float*)d_out;
  gemm_out_kernel<<<1024, 128, 0, stream>>>(
      wrb, attprev + (size_t)BS * NATT, br, out);
  finalize_state_kernel<<<(2*BS*NHID + 255)/256, 256, 0, stream>>>(
      hInt, cInt, out + (size_t)TGTLEN*BS*NOUT,
      out + (size_t)TGTLEN*BS*NOUT + (size_t)2*BS*NHID);
}
